// QCNN_classifier_9620726743441
// MI455X (gfx1250) — matrix operand order repaired and checked by reading, not compiled, not run on hardware
//
#include <hip/hip_runtime.h>

typedef __attribute__((ext_vector_type(8)))  float  v8f;
typedef __attribute__((ext_vector_type(8)))  __bf16 v8bf;
typedef __attribute__((ext_vector_type(16))) __bf16 v16bf;

#define NQ    10
#define DIM   1024
#define BM    64          // rows per block
#define KB    32          // K chunk (one bf16 WMMA K)
#define ARS   40          // A-tile LDS row stride in bf16 (80B: 16B-aligned, conflict-free)
#define NTT   64          // total 16-column tiles (1024/16)
#define NPART 64          // partial sums per row (one per column tile)

// ===================== bf16 split helpers =====================
__device__ __forceinline__ void bsplit(float x, __bf16& h, __bf16& l) {
    h = (__bf16)x;                     // RNE fp32 -> bf16
    l = (__bf16)(x - (float)h);        // residual, ~16 effective mantissa bits total
}
__device__ __forceinline__ unsigned int pack2(__bf16 a, __bf16 b) {
    union { __bf16 h[2]; unsigned int u; } u;
    u.h[0] = a; u.h[1] = b;
    return u.u;
}
// A fragment: two contiguous 16B LDS chunks -> v16bf (contiguous k = 16h + s order, same as B)
__device__ __forceinline__ v16bf ldA(const __bf16* p) {
    v8bf c0 = *(const v8bf*)p;          // K = base .. base+7
    v8bf c1 = *(const v8bf*)(p + 8);    // K = base+8 .. base+15
    return __builtin_shufflevector(c0, c1, 0,1,2,3,4,5,6,7,8,9,10,11,12,13,14,15);
}

// ===================== complex helpers =====================
__device__ __forceinline__ float2 cmul(float2 a, float2 b) {
    return make_float2(fmaf(a.x, b.x, -a.y * b.y), fmaf(a.x, b.y, a.y * b.x));
}
__device__ __forceinline__ float2 cadd(float2 a, float2 b) {
    return make_float2(a.x + b.x, a.y + b.y);
}

// ========== state-vector gates: 1024 amplitudes in LDS, 512 threads ==========
__device__ __forceinline__ void g1(float2* st, int bit,
                                   float2 m00, float2 m01, float2 m10, float2 m11) {
    int t = threadIdx.x;
    int mask = (1 << bit) - 1;
    int i0 = ((t & ~mask) << 1) | (t & mask);
    int i1 = i0 | (1 << bit);
    float2 a = st[i0], b = st[i1];
    st[i0] = cadd(cmul(m00, a), cmul(m01, b));
    st[i1] = cadd(cmul(m10, a), cmul(m11, b));
    __syncthreads();
}
__device__ __forceinline__ void cg1(float2* st, int cbit, int tbit,
                                    float2 m00, float2 m01, float2 m10, float2 m11) {
    int t = threadIdx.x;
    if (t < 256) {
        int lo = cbit < tbit ? cbit : tbit;
        int hb = cbit < tbit ? tbit : cbit;
        int mlo = (1 << lo) - 1, mhb = (1 << hb) - 1;
        int i = ((t & ~mlo) << 1) | (t & mlo);
        i = ((i & ~mhb) << 1) | (i & mhb);
        int i0 = i | (1 << cbit);
        int i1 = i0 | (1 << tbit);
        float2 a = st[i0], b = st[i1];
        st[i0] = cadd(cmul(m00, a), cmul(m01, b));
        st[i1] = cadd(cmul(m10, a), cmul(m11, b));
    }
    __syncthreads();
}

__device__ __forceinline__ void apply_u3(float2* st, int w, float th, float ph, float dl) {
    float s, c;  sincosf(0.5f * th, &s, &c);
    float sp, cp; sincosf(ph, &sp, &cp);
    float sd, cd; sincosf(dl, &sd, &cd);
    float2 ep  = make_float2(cp, sp);
    float2 ed  = make_float2(cd, sd);
    float2 epd = cmul(ep, ed);
    g1(st, (NQ - 1) - w,
       make_float2(c, 0.f),
       make_float2(-ed.x * s, -ed.y * s),
       make_float2(ep.x * s, ep.y * s),
       make_float2(epd.x * c, epd.y * c));
}
__device__ __forceinline__ void apply_ry(float2* st, int w, float th) {
    float s, c; sincosf(0.5f * th, &s, &c);
    g1(st, (NQ - 1) - w, make_float2(c, 0.f), make_float2(-s, 0.f),
                         make_float2(s, 0.f), make_float2(c, 0.f));
}
__device__ __forceinline__ void apply_rz(float2* st, int w, float th) {
    float s, c; sincosf(0.5f * th, &s, &c);
    g1(st, (NQ - 1) - w, make_float2(c, -s), make_float2(0.f, 0.f),
                         make_float2(0.f, 0.f), make_float2(c, s));
}
__device__ __forceinline__ void apply_rx(float2* st, int w, float th) {
    float s, c; sincosf(0.5f * th, &s, &c);
    g1(st, (NQ - 1) - w, make_float2(c, 0.f), make_float2(0.f, -s),
                         make_float2(0.f, -s), make_float2(c, 0.f));
}
__device__ __forceinline__ void apply_x(float2* st, int w) {
    g1(st, (NQ - 1) - w, make_float2(0.f, 0.f), make_float2(1.f, 0.f),
                         make_float2(1.f, 0.f), make_float2(0.f, 0.f));
}
__device__ __forceinline__ void cnot(float2* st, int a, int b) {
    cg1(st, (NQ - 1) - a, (NQ - 1) - b,
        make_float2(0.f, 0.f), make_float2(1.f, 0.f),
        make_float2(1.f, 0.f), make_float2(0.f, 0.f));
}
__device__ __forceinline__ void crz(float2* st, int c, int t, float th) {
    float s, cc; sincosf(0.5f * th, &s, &cc);
    cg1(st, (NQ - 1) - c, (NQ - 1) - t,
        make_float2(cc, -s), make_float2(0.f, 0.f),
        make_float2(0.f, 0.f), make_float2(cc, s));
}
__device__ __forceinline__ void crx(float2* st, int c, int t, float th) {
    float s, cc; sincosf(0.5f * th, &s, &cc);
    cg1(st, (NQ - 1) - c, (NQ - 1) - t,
        make_float2(cc, 0.f), make_float2(0.f, -s),
        make_float2(0.f, -s), make_float2(cc, 0.f));
}

__device__ void conv_block(float2* st, const float* w, int a, int b) {
    apply_u3(st, a, w[0], w[1], w[2]);
    apply_u3(st, b, w[3], w[4], w[5]);
    cnot(st, a, b);
    apply_ry(st, a, w[6]);
    apply_rz(st, b, w[7]);
    cnot(st, b, a);
    apply_ry(st, a, w[8]);
    cnot(st, a, b);
    apply_u3(st, a, w[9],  w[10], w[11]);
    apply_u3(st, b, w[12], w[13], w[14]);
}
__device__ void pool_block(float2* st, const float* p, int c, int t) {
    crz(st, c, t, p[0]);
    apply_x(st, c);
    crx(st, c, t, p[1]);
}

// ===================== kernel 1: per-row squared norms =====================
__global__ __launch_bounds__(256) void row_norm2(const float* __restrict__ x,
                                                 float* __restrict__ nrm2) {
    __shared__ float red[256];
    const int b = blockIdx.x;
    float s = 0.f;
    for (int i = threadIdx.x; i < DIM; i += 256) {
        float v = x[(size_t)b * DIM + i];
        s = fmaf(v, v, s);
    }
    red[threadIdx.x] = s;
    __syncthreads();
    for (int off = 128; off > 0; off >>= 1) {
        if (threadIdx.x < off) red[threadIdx.x] += red[threadIdx.x + off];
        __syncthreads();
    }
    if (threadIdx.x == 0) nrm2[b] = red[0];
}

// ====== kernel 2: build U columns, scatter bf16 hi/lo into WMMA B-fragment order ======
// B-fragment (32x16 bf16, wave32): frag id = (k>>5)*NTT + (n>>4);
//   lane = (n&15) | (((k>>4)&1) << 4); lane holds 16 contiguous K bf16; elem = k&15.
__global__ __launch_bounds__(512) void build_unitary(
        const float* __restrict__ wc1, const float* __restrict__ wc2,
        const float* __restrict__ wp1, const float* __restrict__ wp2,
        const float* __restrict__ wfc,
        __bf16* __restrict__ Bhr, __bf16* __restrict__ Blr,
        __bf16* __restrict__ Bhi, __bf16* __restrict__ Bli) {
    __shared__ float2 st[DIM];
    const int j = blockIdx.x;                 // K index (contraction dim)
    for (int i = threadIdx.x; i < DIM; i += 512)
        st[i] = make_float2(i == j ? 1.f : 0.f, 0.f);
    __syncthreads();

    for (int i = 0; i < NQ; i += 2)     conv_block(st, wc1 + i * 15, i, i + 1);
    for (int i = 1; i < NQ - 1; i += 2) conv_block(st, wc1 + i * 15, i, i + 1);
    conv_block(st, wc1 + 7 * 15, 0, NQ - 1);
    { int idx = 0; for (int i = 0; i < NQ; i += 2, ++idx) pool_block(st, wp1 + idx * 2, i + 1, i); }
    { int idx = 0; for (int i = 0; i < NQ - 2; i += 2, ++idx) conv_block(st, wc2 + idx * 15, i, i + 2); }
    { int idx = 0; for (int i = 0; i < NQ - 2; i += 4, ++idx) pool_block(st, wp2 + idx * 2, i + 2, i); }
    cnot(st, 0, 4); cnot(st, 2, 4); cnot(st, 4, 0);
    apply_rx(st, 0, wfc[0]);
    apply_rx(st, 2, wfc[1]);
    apply_rx(st, 4, wfc[2]);

    const int a  = j >> 5;
    const int hh = (j >> 4) & 1;
    const int q  = j & 15;
    for (int n = threadIdx.x; n < DIM; n += 512) {
        size_t off = ((size_t)(a * NTT + (n >> 4)) * 32 + ((n & 15) | (hh << 4))) * 16 + q;
        __bf16 h, lo;
        bsplit(st[n].x, h, lo); Bhr[off] = h; Blr[off] = lo;
        bsplit(st[n].y, h, lo); Bhi[off] = h; Bli[off] = lo;
    }
}

// === kernel 3: split-bf16 dual GEMM (3 products each) + sign-weighted |y|^2 reduce ===
__global__ __launch_bounds__(128) void gemm_zexp(
        const float*  __restrict__ X,
        const __bf16* __restrict__ Bhr, const __bf16* __restrict__ Blr,
        const __bf16* __restrict__ Bhi, const __bf16* __restrict__ Bli,
        float* __restrict__ part0, float* __restrict__ part2) {
    __shared__ __align__(16) __bf16 XhS[BM * ARS];
    __shared__ __align__(16) __bf16 XlS[BM * ARS];

    const int tid  = threadIdx.x;
    const int lane = tid & 31;
    const int wave = tid >> 5;                 // wave owns 16 columns, all 4 M strips
    const int l    = lane & 15;
    const int hi   = lane >> 4;
    const int mbase = blockIdx.x * BM;
    const int ntg   = blockIdx.y * 4 + wave;   // global 16-column tile id (0..63)

    v8f accr[4], acci[4];
    #pragma unroll
    for (int s = 0; s < 4; ++s) { accr[s] = {}; acci[s] = {}; }

    const int srow = tid >> 1;                 // staging: row 0..63
    const int skh  = (tid & 1) << 4;           // staging: k base 0 or 16

    for (int kc = 0; kc < DIM; kc += KB) {
        __syncthreads();
        {   // stage X tile, splitting fp32 -> bf16 hi/lo, packed b32 LDS stores
            const float* src = X + (size_t)(mbase + srow) * DIM + kc + skh;
            unsigned int* dh = (unsigned int*)&XhS[srow * ARS + skh];
            unsigned int* dl = (unsigned int*)&XlS[srow * ARS + skh];
            #pragma unroll
            for (int q4 = 0; q4 < 4; ++q4) {
                float4 v = ((const float4*)src)[q4];
                __bf16 h0, l0, h1, l1, h2, l2, h3, l3;
                bsplit(v.x, h0, l0); bsplit(v.y, h1, l1);
                bsplit(v.z, h2, l2); bsplit(v.w, h3, l3);
                dh[q4 * 2 + 0] = pack2(h0, h1); dh[q4 * 2 + 1] = pack2(h2, h3);
                dl[q4 * 2 + 0] = pack2(l0, l1); dl[q4 * 2 + 1] = pack2(l2, l3);
            }
        }
        __syncthreads();

        // B fragments: one 32B vector load each, straight from L2-resident arrays
        const size_t boff = ((size_t)((kc >> 5) * NTT + ntg)) * 512 + lane * 16;
        v16bf fhr = *(const v16bf*)(Bhr + boff);
        v16bf flr = *(const v16bf*)(Blr + boff);
        v16bf fhi = *(const v16bf*)(Bhi + boff);
        v16bf fli = *(const v16bf*)(Bli + boff);

        #pragma unroll
        for (int s = 0; s < 4; ++s) {
            const int row = s * 16 + l;
            v16bf ah = ldA(&XhS[row * ARS + hi * 16]);
            v16bf al = ldA(&XlS[row * ARS + hi * 16]);
            accr[s] = __builtin_amdgcn_wmma_f32_16x16x32_bf16(false, ah, false, fhr, (short)0, accr[s], false, false);
            accr[s] = __builtin_amdgcn_wmma_f32_16x16x32_bf16(false, ah, false, flr, (short)0, accr[s], false, false);
            accr[s] = __builtin_amdgcn_wmma_f32_16x16x32_bf16(false, al, false, fhr, (short)0, accr[s], false, false);
            acci[s] = __builtin_amdgcn_wmma_f32_16x16x32_bf16(false, ah, false, fhi, (short)0, acci[s], false, false);
            acci[s] = __builtin_amdgcn_wmma_f32_16x16x32_bf16(false, ah, false, fli, (short)0, acci[s], false, false);
            acci[s] = __builtin_amdgcn_wmma_f32_16x16x32_bf16(false, al, false, fhi, (short)0, acci[s], false, false);
        }
    }

    // epilogue: |y|^2, Z-sign from column index bits (bit9 -> wire0, bit7 -> wire2)
    const int n  = ntg * 16 + l;
    const float s0 = ((n >> 9) & 1) ? -1.f : 1.f;
    const float s2 = ((n >> 7) & 1) ? -1.f : 1.f;
    #pragma unroll
    for (int s = 0; s < 4; ++s) {
        float r0[8], r2[8];
        #pragma unroll
        for (int v = 0; v < 8; ++v) {
            float pr = accr[s][v], pi = acci[s][v];
            float val = fmaf(pr, pr, pi * pi);
            r0[v] = s0 * val;
            r2[v] = s2 * val;
        }
        #pragma unroll
        for (int v = 0; v < 8; ++v) {
            #pragma unroll
            for (int off = 1; off < 16; off <<= 1) {   // xor 1,2,4,8: stays in 16-lane half
                r0[v] += __shfl_xor(r0[v], off, 32);
                r2[v] += __shfl_xor(r2[v], off, 32);
            }
        }
        if (l == 0) {
            #pragma unroll
            for (int v = 0; v < 8; ++v) {
                int b = mbase + s * 16 + v + (hi << 3);
                part0[(size_t)b * NPART + ntg] = r0[v];
                part2[(size_t)b * NPART + ntg] = r2[v];
            }
        }
    }
}

// ============== kernel 4: combine partials, divide by ||x||^2 ==============
__global__ __launch_bounds__(256) void finalize(
        const float* __restrict__ part0, const float* __restrict__ part2,
        const float* __restrict__ nrm2, float* __restrict__ out, int Btot) {
    int b = blockIdx.x * 256 + threadIdx.x;
    if (b < Btot) {
        float s0 = 0.f, s2 = 0.f;
        for (int q = 0; q < NPART; ++q) {
            s0 += part0[(size_t)b * NPART + q];
            s2 += part2[(size_t)b * NPART + q];
        }
        float inv = 1.f / nrm2[b];
        out[b * 2 + 0] = s0 * inv;
        out[b * 2 + 1] = s2 * inv;
    }
}

extern "C" void kernel_launch(void* const* d_in, const int* in_sizes, int n_in,
                              void* d_out, int out_size, void* d_ws, size_t ws_size,
                              hipStream_t stream) {
    (void)n_in; (void)out_size; (void)ws_size;
    const float* x   = (const float*)d_in[0];
    const float* wc1 = (const float*)d_in[1];   // (10,15)
    const float* wc2 = (const float*)d_in[2];   // (4,15)
    const float* wp1 = (const float*)d_in[3];   // (5,2)
    const float* wp2 = (const float*)d_in[4];   // (2,2)
    const float* wfc = (const float*)d_in[5];   // (3,)
    float* out = (float*)d_out;                 // (B,2) fp32

    const int B = in_sizes[0] / DIM;            // 16384

    // ws: Bhr|Blr|Bhi|Bli (2MB each, fragment-ordered bf16) | nrm2 | part0 | part2
    __bf16* Bhr  = (__bf16*)d_ws;
    __bf16* Blr  = Bhr + (size_t)DIM * DIM;
    __bf16* Bhi  = Blr + (size_t)DIM * DIM;
    __bf16* Bli  = Bhi + (size_t)DIM * DIM;
    float*  nrm2 = (float*)(Bli + (size_t)DIM * DIM);
    float*  part0 = nrm2 + B;
    float*  part2 = part0 + (size_t)B * NPART;

    row_norm2<<<B, 256, 0, stream>>>(x, nrm2);
    build_unitary<<<DIM, 512, 0, stream>>>(wc1, wc2, wp1, wp2, wfc, Bhr, Blr, Bhi, Bli);
    gemm_zexp<<<dim3(B / BM, DIM / BM), 128, 0, stream>>>(x, Bhr, Blr, Bhi, Bli, part0, part2);
    finalize<<<(B + 255) / 256, 256, 0, stream>>>(part0, part2, nrm2, out, B);
}
